// ChebGCN3Multi_softmax_63024350101696
// MI455X (gfx1250) — matrix operand order repaired and checked by reading, not compiled, not run on hardware
//
#include <hip/hip_runtime.h>

#define TF 128
#define TC 8
#define GN_EPS 1e-5f
#define GEMM_ROWS 32   // rows per workgroup (2 x 16-row subtiles per wave)

typedef __attribute__((ext_vector_type(16))) __bf16 v16bf;
typedef __attribute__((ext_vector_type(8)))  float  v8f;

union BfFrag { v16bf v; unsigned short u[16]; unsigned int d[8]; };

__device__ __forceinline__ unsigned short f2bf(float x) {
  unsigned u = __float_as_uint(x);
  u = (u + 0x7FFFu + ((u >> 16) & 1u)) >> 16;   // round-to-nearest-even
  return (unsigned short)u;
}

// ---------------------------------------------------------------------------
// O[N,128] = sum_{c=0..3} Tc[N,128] @ W[c] + bias      (bf16 WMMA, f32 accum)
// block = 256 threads = 8 wave32, 32 output rows; wave w -> cols [16w,16w+16)
// for 2 row-subtiles => 32 v_wmma per wave.
// Weights (bf16, [c][out][in]) are DMA'd global->LDS with
// global_load_async_to_lds_b128 (ASYNCcnt), overlapping the fp32->bf16 A stage.
// ---------------------------------------------------------------------------
__global__ void __launch_bounds__(256) cheb_gemm_wmma(
    const float* __restrict__ T0, const float* __restrict__ T1,
    const float* __restrict__ T2, const float* __restrict__ T3,
    const unsigned short* __restrict__ Wl,    // layer weight block, 4*128*128 bf16
    const float* __restrict__ bias,
    float* __restrict__ Out, int Nn) {
  __shared__ __attribute__((aligned(64))) unsigned short ldsW[4 * TF * TF];      // 128 KB
  __shared__ __attribute__((aligned(64))) unsigned short ldsA[4][GEMM_ROWS][TF]; // 32 KB
  const int row0 = blockIdx.x * GEMM_ROWS;

  // 1) async DMA of the whole layer weight block into LDS (no conversion needed)
  {
    const unsigned wbase = (unsigned)(uintptr_t)&ldsW[0];
    for (int i = threadIdx.x * 16; i < 4 * TF * TF * 2; i += 256 * 16) {
      unsigned lds_addr = wbase + (unsigned)i;
      asm volatile("global_load_async_to_lds_b128 %0, %1, %2"
                   :: "v"(lds_addr), "v"((unsigned)i), "s"(Wl)
                   : "memory");
    }
  }

  // 2) meanwhile stage + convert the 4 Chebyshev A-tiles (fp32 -> bf16)
  for (int i = threadIdx.x; i < 4 * GEMM_ROWS * (TF / 4); i += 256) {
    int c = i >> 10;                 // / (GEMM_ROWS*32)
    int r = (i >> 5) & (GEMM_ROWS - 1);
    int q = i & 31;                  // float4 index within row
    const float* Tp = (c == 0) ? T0 : (c == 1) ? T1 : (c == 2) ? T2 : T3;
    int row = row0 + r;
    float4 x = make_float4(0.f, 0.f, 0.f, 0.f);
    if (row < Nn) x = *((const float4*)(Tp + (size_t)row * TF) + q);
    ushort4 b;
    b.x = f2bf(x.x); b.y = f2bf(x.y); b.z = f2bf(x.z); b.w = f2bf(x.w);
    *(ushort4*)&ldsA[c][r][q * 4] = b;
  }

  asm volatile("s_wait_asynccnt 0" ::: "memory");
  __syncthreads();

  const int lane = threadIdx.x & 31;
  const int wave = threadIdx.x >> 5;
  const int m    = lane & 15;
  const int hi   = lane >> 4;        // lane half
  const int col0 = wave * 16;

  v8f acc0 = {}, acc1 = {};
#pragma unroll
  for (int c = 0; c < 4; ++c) {
#pragma unroll
    for (int kk = 0; kk < 4; ++kk) {
      // B (32x16 bf16): N = lane%16, K = half*16 + 0..15 (contiguous in ldsW)
      BfFrag b;
      b.v = *(const v16bf*)&ldsW[((c * TF + (col0 + m)) * TF) + kk * 32 + hi * 16];
      // A (16x32 bf16): half 0 -> K {0..15}, half 1 -> K {16..31}
      BfFrag a0, a1;
#pragma unroll
      for (int v = 0; v < 8; ++v) {
        int kbase = kk * 32 + hi * 16 + v * 2;
        a0.d[v] = *(const unsigned int*)&ldsA[c][m][kbase];
        a1.d[v] = *(const unsigned int*)&ldsA[c][16 + m][kbase];
      }
      acc0 = __builtin_amdgcn_wmma_f32_16x16x32_bf16(
          false, a0.v, false, b.v, (short)0, acc0, false, false);
      acc1 = __builtin_amdgcn_wmma_f32_16x16x32_bf16(
          false, a1.v, false, b.v, (short)0, acc1, false, false);
    }
  }

  const float bc = bias[col0 + m];
#pragma unroll
  for (int r = 0; r < 8; ++r) {              // C/D: VGPR r -> row r + half*8
    int row = row0 + hi * 8 + r;
    if (row < Nn) Out[(size_t)row * TF + col0 + m] = acc0[r] + bc;
    int row2 = row + 16;
    if (row2 < Nn) Out[(size_t)row2 * TF + col0 + m] = acc1[r] + bc;
  }
}

// ---------------------------------------------------------------------------
// Sparse pieces (L2-resident: node matrix = 25.6 MB << 192 MB L2)
// ---------------------------------------------------------------------------
__global__ void __launch_bounds__(256) spmm_scatter(
    const int* __restrict__ src, const int* __restrict__ dst,
    const float* __restrict__ ew, const float* __restrict__ X,
    float* __restrict__ Y, float scale, int Ee) {
  int t = blockIdx.x * 256 + threadIdx.x;
  int e = t >> 5;
  if (e >= Ee) return;
  int lane = t & 31;
  int s = src[e], d = dst[e];
  float w = ew[e] * scale;
  int e2 = e + 4096;
  if (e2 < Ee) __builtin_prefetch(X + (size_t)src[e2] * TF + lane * 4, 0, 1);
  const float4 x = *((const float4*)(X + (size_t)s * TF) + lane);
  float* yp = Y + (size_t)d * TF + lane * 4;
  atomicAdd(yp + 0, w * x.x);
  atomicAdd(yp + 1, w * x.y);
  atomicAdd(yp + 2, w * x.z);
  atomicAdd(yp + 3, w * x.w);
}

__global__ void zero_f32(float* __restrict__ p, int n) {
  int i = blockIdx.x * 256 + threadIdx.x;
  if (i < n) p[i] = 0.f;
}

__global__ void scale_copy(const float* __restrict__ x, float* __restrict__ y,
                           float a, int n) {
  int i = blockIdx.x * 256 + threadIdx.x;
  if (i < n) y[i] = a * x[i];
}

__global__ void deg_accum(const int* __restrict__ src, float* __restrict__ deg, int Ee) {
  int i = blockIdx.x * 256 + threadIdx.x;
  if (i < Ee) atomicAdd(&deg[src[i]], 1.f);
}

__global__ void deg_to_dis(float* __restrict__ deg, int Nn) {
  int i = blockIdx.x * 256 + threadIdx.x;
  if (i < Nn) { float d = deg[i]; deg[i] = d > 0.f ? rsqrtf(d) : 0.f; }
}

__global__ void edge_weights(const int* __restrict__ src, const int* __restrict__ dst,
                             const float* __restrict__ dis, float* __restrict__ ew, int Ee) {
  int i = blockIdx.x * 256 + threadIdx.x;
  if (i < Ee) ew[i] = -dis[src[i]] * dis[dst[i]];
}

// ---------------------------------------------------------------------------
// GraphNorm: one sum/sumsq pass, var = E[x^2] - ms(2-ms)*mean^2
// ---------------------------------------------------------------------------
__global__ void __launch_bounds__(256) gn_stats(const float* __restrict__ X,
                                                float* __restrict__ sums, int Nn) {
  __shared__ float s1[256], s2[256];
  int f = threadIdx.x & (TF - 1);
  int g = threadIdx.x >> 7;
  float a = 0.f, q = 0.f;
  for (int r = blockIdx.x * 2 + g; r < Nn; r += gridDim.x * 2) {
    float v = X[(size_t)r * TF + f];
    a += v; q += v * v;
  }
  s1[threadIdx.x] = a; s2[threadIdx.x] = q;
  __syncthreads();
  if (g == 0) {
    atomicAdd(&sums[f],      s1[threadIdx.x] + s1[threadIdx.x + 128]);
    atomicAdd(&sums[TF + f], s2[threadIdx.x] + s2[threadIdx.x + 128]);
  }
}

__global__ void gn_finalize(const float* __restrict__ sums, const float* __restrict__ ms,
                            float* __restrict__ mean, float* __restrict__ istd, int Nn) {
  int f = threadIdx.x;
  float mu = sums[f] / (float)Nn;
  float m2 = sums[TF + f] / (float)Nn;
  float s  = ms[f];
  float var = m2 - s * (2.f - s) * mu * mu;
  mean[f] = mu;
  istd[f] = rsqrtf(var + GN_EPS);
}

__global__ void gn_apply(const float* __restrict__ X, const float* __restrict__ X0,
                         const float* __restrict__ gw, const float* __restrict__ gb,
                         const float* __restrict__ ms, const float* __restrict__ mean,
                         const float* __restrict__ istd, float* __restrict__ H,
                         int total, int residual) {
  int i = blockIdx.x * 256 + threadIdx.x;
  if (i >= total) return;
  int f = i & (TF - 1);
  float x = X[i];
  float v = gw[f] * (x - ms[f] * mean[f]) * istd[f] + gb[f];
  float h;
  if (residual) { float t = X0[i] + v; h = t > 0.f ? t : 0.f; }
  else          { h = v > 0.f ? v : 0.1f * v; }
  H[i] = h;
}

// ---------------------------------------------------------------------------
// Head: pooled = relu(colmean), logits = pooled @ lw^T + lb, softmax
// ---------------------------------------------------------------------------
__global__ void head_kernel(const float* __restrict__ sums, const float* __restrict__ lw,
                            const float* __restrict__ lb, float* __restrict__ out, int Nn) {
  __shared__ float pooled[TF];
  __shared__ float partial[TC][32];
  __shared__ float logits[TC];
  int t = threadIdx.x;
  if (t < TF) { float p = sums[t] / (float)Nn; pooled[t] = p > 0.f ? p : 0.f; }
  __syncthreads();
  int c = t >> 5, lane = t & 31;
  float p = 0.f;
  for (int f = lane; f < TF; f += 32) p += pooled[f] * lw[c * TF + f];
  partial[c][lane] = p;
  __syncthreads();
  if (t < TC) { float s = lb[t]; for (int i = 0; i < 32; ++i) s += partial[t][i]; logits[t] = s; }
  __syncthreads();
  if (t == 0) {
    float mx = logits[0];
    for (int i = 1; i < TC; ++i) mx = fmaxf(mx, logits[i]);
    float se = 0.f, e[TC];
    for (int i = 0; i < TC; ++i) { e[i] = __expf(logits[i] - mx); se += e[i]; }
    for (int i = 0; i < TC; ++i) out[i] = e[i] / se;
  }
}

// convs_w slice [4][4][128][128] (l,k,in,out) fp32 -> Wbt [4][4][128][128] (l,k,out,in) bf16
__global__ void convert_weights(const float* __restrict__ W,
                                unsigned short* __restrict__ Wbt, int total) {
  int i = blockIdx.x * 256 + threadIdx.x;
  if (i >= total) return;
  int lk  = i >> 14;
  int rem = i & 16383;
  int o   = rem >> 7;
  int in  = rem & 127;
  Wbt[i] = f2bf(W[((size_t)lk << 14) + ((size_t)in << 7) + o]);
}

// ---------------------------------------------------------------------------
extern "C" void kernel_launch(void* const* d_in, const int* in_sizes, int n_in,
                              void* d_out, int out_size, void* d_ws, size_t ws_size,
                              hipStream_t stream) {
  (void)n_in; (void)out_size; (void)ws_size;
  const int E = in_sizes[0] / 2;
  const int N = in_sizes[1] / TF;
  const int* src = (const int*)d_in[0];
  const int* dst = src + E;
  const float* feats[3] = {(const float*)d_in[1], (const float*)d_in[2], (const float*)d_in[3]};
  const float* convs_w = (const float*)d_in[4];
  const float* convs_b = (const float*)d_in[5];
  const float* gn_w  = (const float*)d_in[6];
  const float* gn_b  = (const float*)d_in[7];
  const float* gn_ms = (const float*)d_in[8];
  const float* lin_w = (const float*)d_in[9];
  const float* lin_b = (const float*)d_in[10];
  float* out = (float*)d_out;

  char* ws = (char*)d_ws;
  size_t off = 0;
  auto alloc = [&](size_t bytes) -> void* {
    off = (off + 255) & ~(size_t)255;
    void* p = ws + off;
    off += bytes;
    return p;
  };
  const size_t NF = (size_t)N * TF;
  float* dis = (float*)alloc((size_t)N * 4);
  float* ew  = (float*)alloc((size_t)E * 4);
  float* T1  = (float*)alloc(NF * 4);
  float* T2  = (float*)alloc(NF * 4);
  float* T3  = (float*)alloc(NF * 4);
  float* H   = (float*)alloc(NF * 4);
  float* O   = (float*)alloc(NF * 4);
  unsigned short* Wbt = (unsigned short*)alloc((size_t)4 * 4 * TF * TF * 2);
  float* sums = (float*)alloc(2 * TF * 4);
  float* mean = (float*)alloc(TF * 4);
  float* istd = (float*)alloc(TF * 4);

  int gN    = (N + 255) / 256;
  int gNF   = (int)((NF + 255) / 256);
  int gE    = (E + 255) / 256;
  int gE32  = (int)(((size_t)E * 32 + 255) / 256);
  int gTile = (N + GEMM_ROWS - 1) / GEMM_ROWS;
  int wTot  = 4 * 4 * TF * TF;

  // sym-normalized edge weights: ew = -dis[src]*dis[dst]
  zero_f32<<<gN, 256, 0, stream>>>(dis, N);
  deg_accum<<<gE, 256, 0, stream>>>(src, dis, E);
  deg_to_dis<<<gN, 256, 0, stream>>>(dis, N);
  edge_weights<<<gE, 256, 0, stream>>>(src, dst, dis, ew, E);

  for (int b = 0; b < 3; ++b) {
    convert_weights<<<(wTot + 255) / 256, 256, 0, stream>>>(
        convs_w + (size_t)b * wTot, Wbt, wTot);
    scale_copy<<<gNF, 256, 0, stream>>>(feats[b], H, 1.0f, (int)NF);

    for (int l = 0; l < 4; ++l) {
      const size_t pi = (size_t)b * 4 + l;   // per-(branch,layer) param index
      // Chebyshev recursion: T1 = Lhat(H); T2 = 2*Lhat(T1) - H; T3 = 2*Lhat(T2) - T1
      zero_f32<<<gNF, 256, 0, stream>>>(T1, (int)NF);
      spmm_scatter<<<gE32, 256, 0, stream>>>(src, dst, ew, H, T1, 1.0f, E);
      scale_copy<<<gNF, 256, 0, stream>>>(H, T2, -1.0f, (int)NF);
      spmm_scatter<<<gE32, 256, 0, stream>>>(src, dst, ew, T1, T2, 2.0f, E);
      scale_copy<<<gNF, 256, 0, stream>>>(T1, T3, -1.0f, (int)NF);
      spmm_scatter<<<gE32, 256, 0, stream>>>(src, dst, ew, T2, T3, 2.0f, E);

      cheb_gemm_wmma<<<gTile, 256, 0, stream>>>(
          H, T1, T2, T3, Wbt + (size_t)l * 4 * TF * TF, convs_b + pi * TF, O, N);

      zero_f32<<<1, 256, 0, stream>>>(sums, 2 * TF);
      gn_stats<<<256, 256, 0, stream>>>(O, sums, N);
      gn_finalize<<<1, TF, 0, stream>>>(sums, gn_ms + pi * TF, mean, istd, N);
      gn_apply<<<gNF, 256, 0, stream>>>(O, feats[b], gn_w + pi * TF, gn_b + pi * TF,
                                        gn_ms + pi * TF, mean, istd, H, (int)NF,
                                        l == 3 ? 1 : 0);
    }
    zero_f32<<<1, 256, 0, stream>>>(sums, 2 * TF);
    gn_stats<<<256, 256, 0, stream>>>(H, sums, N);
    head_kernel<<<1, 256, 0, stream>>>(sums, lin_w + (size_t)b * TC * TF,
                                       lin_b + (size_t)b * TC, out + (size_t)b * TC, N);
  }
}
